// DrugGINEncoder_69441031241771
// MI455X (gfx1250) — compile-verified, matrix operand order checked by reading, not run on hardware
//
#include <hip/hip_runtime.h>

#define HD 256
#define NODES 200000
#define EDGES 400000
#define NB 4096
#define NLAY 5
#define BN_EPS 1e-5f

typedef __attribute__((ext_vector_type(16))) __bf16 v16bf;
typedef __attribute__((ext_vector_type(8)))  __bf16 v8bf;
typedef __attribute__((ext_vector_type(8)))  float  v8f;

// ---------------------------------------------------------------------------
// small helpers
// ---------------------------------------------------------------------------
static __device__ __forceinline__ v16bf pack16(const __bf16* lo, const __bf16* hi) {
  v8bf a = *(const v8bf*)lo;
  v8bf b = *(const v8bf*)hi;
  v16bf r;
#pragma unroll
  for (int i = 0; i < 8; ++i) { r[i] = a[i]; r[i + 8] = b[i]; }
  return r;
}

__global__ void zero_kernel(float* __restrict__ p, long long n) {
  long long i = (long long)blockIdx.x * blockDim.x + threadIdx.x;
  long long stride = (long long)gridDim.x * blockDim.x;
  for (; i < n; i += stride) p[i] = 0.0f;
}

__global__ void copy_kernel(float* __restrict__ dst, const float* __restrict__ src, long long n) {
  long long i = (long long)blockIdx.x * blockDim.x + threadIdx.x;
  long long stride = (long long)gridDim.x * blockDim.x;
  for (; i < n; i += stride) dst[i] = src[i];
}

__global__ void tobf16_kernel(__bf16* __restrict__ dst, const float* __restrict__ src, long long n) {
  long long i = (long long)blockIdx.x * blockDim.x + threadIdx.x;
  long long stride = (long long)gridDim.x * blockDim.x;
  for (; i < n; i += stride) dst[i] = (__bf16)src[i];
}

// W [256(k), 256(n)] f32 -> Wt [256(n), 256(k)] bf16 (transpose for B-fragment loads)
__global__ void convw_kernel(const float* __restrict__ W, __bf16* __restrict__ Wt) {
  int idx = blockIdx.x * 256 + threadIdx.x;   // 65536 threads, coalesced read
  int k = idx >> 8, n = idx & 255;
  Wt[(size_t)n * HD + k] = (__bf16)W[idx];
}

// ---------------------------------------------------------------------------
// h = x @ W(10x256) + b
// ---------------------------------------------------------------------------
__global__ __launch_bounds__(256) void node_embed_kernel(
    const float* __restrict__ x, const float* __restrict__ W,
    const float* __restrict__ b, float* __restrict__ h, int Nn) {
  __shared__ float w[10 * HD];
  __shared__ float bb[HD];
  int tid = threadIdx.x;
  for (int i = tid; i < 10 * HD; i += 256) w[i] = W[i];
  bb[tid] = b[tid];
  __syncthreads();
  for (int r = blockIdx.x; r < Nn; r += gridDim.x) {
    float acc = bb[tid];
#pragma unroll
    for (int k = 0; k < 10; ++k) acc += x[(size_t)r * 10 + k] * w[k * HD + tid];
    h[(size_t)r * HD + tid] = acc;
  }
}

__global__ void vinit_kernel(float* __restrict__ v, const float* __restrict__ emb) {
  v[(size_t)blockIdx.x * HD + threadIdx.x] = emb[threadIdx.x];
}

// h[r] += v[batch[r]]
__global__ void vadd_kernel(float* __restrict__ h, const float* __restrict__ v,
                            const int* __restrict__ batch) {
  int r = blockIdx.x, c = threadIdx.x;
  h[(size_t)r * HD + c] += v[(size_t)batch[r] * HD + c];
}

// ---------------------------------------------------------------------------
// GINE message: aggr[dst] += relu(h[src] + ea @ We + be)   (one wave per edge)
// ---------------------------------------------------------------------------
__global__ __launch_bounds__(256) void edge_msg_kernel(
    const float* __restrict__ h, const int* __restrict__ src, const int* __restrict__ dst,
    const float* __restrict__ ea, const float* __restrict__ We, const float* __restrict__ be,
    float* __restrict__ aggr, int E) {
  __shared__ float w[7 * HD];
  __shared__ float wb[HD];
  int tid = threadIdx.x;
  for (int i = tid; i < 7 * HD; i += 256) w[i] = We[i];
  wb[tid] = be[tid];
  __syncthreads();
  int e = blockIdx.x * 8 + (tid >> 5);
  if (e >= E) return;
  int lane = tid & 31;
  int s = src[e], d = dst[e];
  float a0 = ea[(size_t)e * 7 + 0], a1 = ea[(size_t)e * 7 + 1], a2 = ea[(size_t)e * 7 + 2];
  float a3 = ea[(size_t)e * 7 + 3], a4 = ea[(size_t)e * 7 + 4], a5 = ea[(size_t)e * 7 + 5];
  float a6 = ea[(size_t)e * 7 + 6];
  int c0 = lane * 8;
  const float* hp = h + (size_t)s * HD + c0;
  float* ap = aggr + (size_t)d * HD + c0;
#pragma unroll
  for (int j = 0; j < 8; ++j) {
    int c = c0 + j;
    float ev = wb[c] + a0 * w[0 * HD + c] + a1 * w[1 * HD + c] + a2 * w[2 * HD + c] +
               a3 * w[3 * HD + c] + a4 * w[4 * HD + c] + a5 * w[5 * HD + c] + a6 * w[6 * HD + c];
    float m = hp[j] + ev;
    m = fmaxf(m, 0.0f);
    atomicAdd(ap + j, m);
  }
}

// abf = bf16((1+eps)*h + aggr)
__global__ void prep_comb_kernel(const float* __restrict__ h, const float* __restrict__ aggr,
                                 const float* __restrict__ eps_l, __bf16* __restrict__ out) {
  float s = 1.0f + *eps_l;
  size_t i = (size_t)blockIdx.x * HD + threadIdx.x;
  out[i] = (__bf16)(s * h[i] + aggr[i]);
}

// column sums / sumsq of z [M,256]
__global__ __launch_bounds__(256) void bn_stats_kernel(const float* __restrict__ z,
                                                       float* __restrict__ stats, int M) {
  int c = threadIdx.x;
  float s = 0.0f, s2 = 0.0f;
  for (int r = blockIdx.x; r < M; r += gridDim.x) {
    float v = z[(size_t)r * HD + c];
    s += v; s2 += v * v;
  }
  atomicAdd(&stats[c], s);
  atomicAdd(&stats[HD + c], s2);
}

// abf = bf16(relu(bn(z)))
__global__ void prep_bnrelu_kernel(const float* __restrict__ z, const float* __restrict__ stats,
                                   float invM, const float* __restrict__ g,
                                   const float* __restrict__ be, __bf16* __restrict__ out) {
  int c = threadIdx.x;
  float mu = stats[c] * invM;
  float var = stats[HD + c] * invM - mu * mu;
  float sc = g[c] * rsqrtf(var + BN_EPS);
  float sh = be[c] - mu * sc;
  size_t i = (size_t)blockIdx.x * HD + c;
  out[i] = (__bf16)fmaxf(z[i] * sc + sh, 0.0f);
}

// h = relu(bn(z)); accum[batch[r]] += h   (segment-sum fusion)
__global__ void apply_kernel(const float* __restrict__ z, const float* __restrict__ stats,
                             float invM, const float* __restrict__ g, const float* __restrict__ be,
                             float* __restrict__ h, const int* __restrict__ batch,
                             float* __restrict__ accum) {
  int c = threadIdx.x, r = blockIdx.x;
  float mu = stats[c] * invM;
  float var = stats[HD + c] * invM - mu * mu;
  float sc = g[c] * rsqrtf(var + BN_EPS);
  float sh = be[c] - mu * sc;
  size_t i = (size_t)r * HD + c;
  float v = fmaxf(z[i] * sc + sh, 0.0f);
  h[i] = v;
  atomicAdd(&accum[(size_t)batch[r] * HD + c], v);
}

// ---------------------------------------------------------------------------
// D[M,256] = A[M,256](bf16) @ W[256,256] + bias,  W given transposed bf16 Wt[n][k]
// block = 8 waves; wave owns a 16-row M tile x all 256 cols; Wt staged in LDS
// in two 64KB halves; K=256 -> 8 x v_wmma_f32_16x16x32_bf16 per 16x16 tile.
// ---------------------------------------------------------------------------
__global__ __launch_bounds__(256) void wmma_gemm_kernel(
    const __bf16* __restrict__ A, const __bf16* __restrict__ Wt,
    const float* __restrict__ bias, float* __restrict__ D, int M) {
  __shared__ __align__(16) __bf16 ldsW[128 * HD];   // 64 KB: half of Wt
  const int tid  = threadIdx.x;
  const int wave = tid >> 5;
  const int lane = tid & 31;
  const int tile = blockIdx.x * 8 + wave;
  const int m0   = tile * 16;
  const bool active = (m0 < M);           // wave-uniform -> EXEC stays all-ones

  const int lh     = lane & 15;
  const int kshift = (lane >> 4) << 3;    // 0 for lanes 0-15, 8 for 16-31

  // cache the 8 A fragments (K = 0..255) for this wave's 16 rows
  v16bf afrag[8];
  if (active) {
    const __bf16* arow = A + (size_t)(m0 + lh) * HD;
#pragma unroll
    for (int ks = 0; ks < 8; ++ks) {
      int base = ks * 32 + kshift;
      afrag[ks] = pack16(arow + base, arow + base + 16);
    }
  }

#pragma unroll
  for (int half = 0; half < 2; ++half) {
    if (half) __syncthreads();            // previous compute done reading LDS
    {                                     // stage 128 output columns of Wt
      const uint4* s = (const uint4*)(Wt + (size_t)half * 128 * HD);
      uint4* d = (uint4*)ldsW;
      for (int i = tid; i < 4096; i += 256) d[i] = s[i];
    }
    __syncthreads();
    if (active) {
#pragma unroll
      for (int ng = 0; ng < 2; ++ng) {    // two 64-column groups per half
        v8f acc[4];
#pragma unroll
        for (int t = 0; t < 4; ++t) acc[t] = (v8f)0.0f;
#pragma unroll
        for (int ks = 0; ks < 8; ++ks) {
          int kb = ks * 32 + kshift;
#pragma unroll
          for (int t = 0; t < 4; ++t) {
            const __bf16* bp = ldsW + (size_t)(ng * 64 + t * 16 + lh) * HD + kb;
            v16bf bfrag = pack16(bp, bp + 16);
            acc[t] = __builtin_amdgcn_wmma_f32_16x16x32_bf16(
                false, afrag[ks], false, bfrag, (short)0, acc[t], false, false);
          }
        }
#pragma unroll
        for (int t = 0; t < 4; ++t) {
          int ncol = half * 128 + ng * 64 + t * 16 + lh;
          float bs = bias[ncol];
#pragma unroll
          for (int r = 0; r < 8; ++r) {
            D[(size_t)(m0 + r + kshift) * HD + ncol] = acc[t][r] + bs;
          }
        }
      }
    }
  }
}

// ---------------------------------------------------------------------------
extern "C" void kernel_launch(void* const* d_in, const int* in_sizes, int n_in,
                              void* d_out, int out_size, void* d_ws, size_t ws_size,
                              hipStream_t stream) {
  const float* x         = (const float*)d_in[0];
  const int*   eidx      = (const int*)d_in[1];     // [2,E]
  const float* eattr     = (const float*)d_in[2];
  const int*   batch     = (const int*)d_in[3];
  const float* node_in_w = (const float*)d_in[4];
  const float* node_in_b = (const float*)d_in[5];
  const float* vn_embed  = (const float*)d_in[6];
  const float* epsp      = (const float*)d_in[7];
  const float* edge_w    = (const float*)d_in[8];
  const float* edge_b    = (const float*)d_in[9];
  const float* mlp_w1    = (const float*)d_in[10];
  const float* mlp_b1    = (const float*)d_in[11];
  const float* mlp_g1    = (const float*)d_in[12];
  const float* mlp_be1   = (const float*)d_in[13];
  const float* mlp_w2    = (const float*)d_in[14];
  const float* mlp_b2    = (const float*)d_in[15];
  const float* bn_g      = (const float*)d_in[16];
  const float* bn_b      = (const float*)d_in[17];
  const float* vn_w1     = (const float*)d_in[18];
  const float* vn_b1     = (const float*)d_in[19];
  const float* vn_g1     = (const float*)d_in[20];
  const float* vn_be1    = (const float*)d_in[21];
  const float* vn_w2     = (const float*)d_in[22];
  const float* vn_b2     = (const float*)d_in[23];

  float* pooled = (float*)d_out;                 // [B,256]
  float* h      = pooled + (size_t)NB * HD;      // [N,256]  (node_feats output)

  char* wp = (char*)d_ws;
  float*  aggr  = (float*)wp;  wp += (size_t)NODES * HD * 4;
  float*  zbuf  = (float*)wp;  wp += (size_t)NODES * HD * 4;
  __bf16* abf   = (__bf16*)wp; wp += (size_t)NODES * HD * 2;
  float*  vbuf  = (float*)wp;  wp += (size_t)NB * HD * 4;
  float*  vsum  = (float*)wp;  wp += (size_t)NB * HD * 4;
  float*  zv    = (float*)wp;  wp += (size_t)NB * HD * 4;
  float*  stats = (float*)wp;  wp += 512 * 4;
  __bf16* wbf   = (__bf16*)wp;                    // 18 x 256x256 bf16, transposed

  // one-time weight conversion (transpose + bf16)
  for (int l = 0; l < NLAY; ++l) {
    convw_kernel<<<256, 256, 0, stream>>>(mlp_w1 + (size_t)l * HD * HD, wbf + (size_t)l * HD * HD);
    convw_kernel<<<256, 256, 0, stream>>>(mlp_w2 + (size_t)l * HD * HD, wbf + (size_t)(5 + l) * HD * HD);
  }
  for (int l = 0; l < NLAY - 1; ++l) {
    convw_kernel<<<256, 256, 0, stream>>>(vn_w1 + (size_t)l * HD * HD, wbf + (size_t)(10 + l) * HD * HD);
    convw_kernel<<<256, 256, 0, stream>>>(vn_w2 + (size_t)l * HD * HD, wbf + (size_t)(14 + l) * HD * HD);
  }

  node_embed_kernel<<<4096, 256, 0, stream>>>(x, node_in_w, node_in_b, h, NODES);
  vinit_kernel<<<NB, 256, 0, stream>>>(vbuf, vn_embed);

  const int gN = (NODES / 16 + 7) / 8;   // 1563 blocks (last partial, wave-guarded)
  const int gB = (NB / 16) / 8;          // 32 blocks

  for (int l = 0; l < NLAY; ++l) {
    vadd_kernel<<<NODES, 256, 0, stream>>>(h, vbuf, batch);
    zero_kernel<<<2048, 256, 0, stream>>>(aggr, (long long)NODES * HD);
    edge_msg_kernel<<<(EDGES + 7) / 8, 256, 0, stream>>>(
        h, eidx, eidx + EDGES, eattr, edge_w + (size_t)l * 7 * HD, edge_b + (size_t)l * HD,
        aggr, EDGES);
    prep_comb_kernel<<<NODES, 256, 0, stream>>>(h, aggr, epsp + l, abf);
    wmma_gemm_kernel<<<gN, 256, 0, stream>>>(abf, wbf + (size_t)l * HD * HD,
                                             mlp_b1 + (size_t)l * HD, zbuf, NODES);
    zero_kernel<<<2, 256, 0, stream>>>(stats, 512);
    bn_stats_kernel<<<512, 256, 0, stream>>>(zbuf, stats, NODES);
    prep_bnrelu_kernel<<<NODES, 256, 0, stream>>>(zbuf, stats, 1.0f / NODES,
                                                  mlp_g1 + (size_t)l * HD,
                                                  mlp_be1 + (size_t)l * HD, abf);
    wmma_gemm_kernel<<<gN, 256, 0, stream>>>(abf, wbf + (size_t)(5 + l) * HD * HD,
                                             mlp_b2 + (size_t)l * HD, zbuf, NODES);
    zero_kernel<<<2, 256, 0, stream>>>(stats, 512);
    bn_stats_kernel<<<512, 256, 0, stream>>>(zbuf, stats, NODES);
    float* accum;
    if (l < NLAY - 1) {
      copy_kernel<<<1024, 256, 0, stream>>>(vsum, vbuf, (long long)NB * HD);
      accum = vsum;
    } else {
      zero_kernel<<<1024, 256, 0, stream>>>(pooled, (long long)NB * HD);
      accum = pooled;
    }
    apply_kernel<<<NODES, 256, 0, stream>>>(zbuf, stats, 1.0f / NODES,
                                            bn_g + (size_t)l * HD, bn_b + (size_t)l * HD,
                                            h, batch, accum);
    if (l < NLAY - 1) {
      tobf16_kernel<<<1024, 256, 0, stream>>>(abf, vsum, (long long)NB * HD);
      wmma_gemm_kernel<<<gB, 256, 0, stream>>>(abf, wbf + (size_t)(10 + l) * HD * HD,
                                               vn_b1 + (size_t)l * HD, zv, NB);
      zero_kernel<<<2, 256, 0, stream>>>(stats, 512);
      bn_stats_kernel<<<256, 256, 0, stream>>>(zv, stats, NB);
      prep_bnrelu_kernel<<<NB, 256, 0, stream>>>(zv, stats, 1.0f / NB,
                                                 vn_g1 + (size_t)l * HD,
                                                 vn_be1 + (size_t)l * HD, abf);
      wmma_gemm_kernel<<<gB, 256, 0, stream>>>(abf, wbf + (size_t)(14 + l) * HD * HD,
                                               vn_b2 + (size_t)l * HD, vbuf, NB);
    }
  }
}
